// AttentionModule_61735859913434
// MI455X (gfx1250) — matrix operand order repaired and checked by reading, not compiled, not run on hardware
//
#include <hip/hip_runtime.h>
#include <hip/hip_bf16.h>

// ---------------------------------------------------------------------------
// MI455X (gfx1250) implementation.
// f16 NHWC activations / [k][cout] weights, v_wmma_f32_16x16x32_f16 (f32 acc).
// Conv weight tiles staged to LDS via the Tensor Data Mover (tensor_load_to_lds
// + s_wait_tensorcnt).  All GEMM-loop loads are branchless: clamped always-valid
// addresses + wide b128 loads + cndmask zeroing.
// ---------------------------------------------------------------------------

typedef __attribute__((ext_vector_type(16))) _Float16 v16h;
typedef __attribute__((ext_vector_type(8)))  float    v8f;
typedef __attribute__((ext_vector_type(4)))  unsigned u32x4;
typedef __attribute__((ext_vector_type(8)))  unsigned u32x8;
typedef _Float16 h16;

union Frag { v16h v; _Float16 h[16]; unsigned u[8]; u32x4 q[2]; };
union G0u  { u32x4 v; unsigned u[4]; };
union G1u  { u32x8 v; unsigned u[8]; };

__device__ __forceinline__ int iclamp(int v, int lo, int hi) {
  return v < lo ? lo : (v > hi ? hi : v);
}

// ======================= generic implicit-GEMM conv ========================
// in  : f16 NHWC [IMG][H][W][CIN]
// wt  : f16 [k][Cout], k = (ky*KS+kx)*CIN + cin
// One block = 4 waves sharing one 16-wide cout tile; weight slice [Ktot x 16]
// TDM-loaded into LDS once per block.  act: 0 none, 1 relu.
template <int CIN, int KS, int DIL>
__global__ __launch_bounds__(128) void conv_wmma(
    const h16* __restrict__ in, const h16* __restrict__ wt,
    const float* __restrict__ bias, const float* __restrict__ res32,
    float* __restrict__ out32, h16* __restrict__ out16,
    int IMG, int H, int W, int Cout, int act, int oStride, int oOff)
{
  constexpr int Ktot  = KS * KS * CIN;
  constexpr int HALFK = KS >> 1;
  __shared__ h16 Bs[Ktot * 16];

  const int lane = threadIdx.x & 31;
  const int wib  = threadIdx.x >> 5;
  const int Wt = W >> 4;
  const int Mtiles = IMG * H * Wt;
  const int Ntiles = Cout >> 4;
  const int nt  = blockIdx.x % Ntiles;
  const int mtg = blockIdx.x / Ntiles;
  const int mt  = mtg * 4 + wib;

  // ---- TDM: stage weight slice [Ktot][16] (cols nt*16..) into LDS ---------
  if (wib == 0) {
    const unsigned long long ga = (unsigned long long)(const void*)(wt + ((size_t)nt << 4));
    G0u g0;
    g0.u[0] = 1u;                                        // count=1, user D#
    g0.u[1] = (unsigned)(size_t)(void*)Bs;               // lds_addr (bytes)
    g0.u[2] = (unsigned)ga;                              // global_addr lo
    g0.u[3] = (unsigned)((ga >> 32) & 0x1ffffffull) | (2u << 30);  // addr hi | type=2
    G1u g1;
    g1.u[0] = 1u << 16;                                  // data_size = 2 bytes
    g1.u[1] = ((unsigned)Cout & 0xffffu) << 16;          // tensor_dim0[15:0]
    g1.u[2] = (((unsigned)Cout >> 16) & 0xffffu)
            | (((unsigned)Ktot & 0xffffu) << 16);        // tensor_dim1[15:0]
    g1.u[3] = (((unsigned)Ktot >> 16) & 0xffffu)
            | (16u << 16);                               // tile_dim0 = 16
    g1.u[4] = (unsigned)Ktot & 0xffffu;                  // tile_dim1 = Ktot
    g1.u[5] = (unsigned)Cout;                            // tensor_dim0_stride lo32
    g1.u[6] = 0u;
    g1.u[7] = 0u;
    asm volatile("tensor_load_to_lds %0, %1" :: "s"(g0.v), "s"(g1.v) : "memory");
    __builtin_amdgcn_s_wait_tensorcnt(0);
  }
  __syncthreads();

  if (mt >= Mtiles) return;                 // wave-uniform: EXEC stays all-1s

  const int xt  = mt % Wt;
  const int row = mt / Wt;
  const int y   = row % H;
  const int img = row / H;
  const int x0  = xt << 4;
  const int hi  = lane >> 4;
  const int l15 = lane & 15;
  const int ax  = x0 + l15;                 // A fragment: lane -> M (pixel)

  __builtin_prefetch(in + (((size_t)img * H + y) * W + ax) * CIN, 0, 1);

  v8f acc = {0.f,0.f,0.f,0.f,0.f,0.f,0.f,0.f};

  for (int k0 = 0; k0 < Ktot; k0 += 32) {
    Frag A, B;
    // A: two groups of 8 consecutive K (CIN%8==0 -> single tap, contiguous cin)
#pragma unroll
    for (int g = 0; g < 2; ++g) {
      const int kg  = k0 + (hi << 3) + (g << 4);
      const int tap = kg / CIN;
      const int cin = kg - tap * CIN;
      const int ky  = tap / KS;
      const int kx  = tap - ky * KS;
      const int yy  = y  + (ky - HALFK) * DIL;
      const int xx  = ax + (kx - HALFK) * DIL;
      const bool ok = ((unsigned)yy < (unsigned)H) && ((unsigned)xx < (unsigned)W)
                      && ((Ktot % 32 == 0) || kg < Ktot);
      const int yc = iclamp(yy, 0, H - 1);
      const int xc = iclamp(xx, 0, W - 1);
      const u32x4 av = *(const u32x4*)(in + ((((size_t)img * H + yc) * W + xc) * CIN + cin));
#pragma unroll
      for (int j = 0; j < 4; ++j) A.u[(g << 2) + j] = ok ? av[j] : 0u;
    }
    // B: element s -> LDS row k (table order, same as A), column l15
    {
#pragma unroll
      for (int s = 0; s < 16; ++s) {
        const int kB  = k0 + (hi << 3) + (s & 7) + ((s >> 3) << 4);
        const bool ok = (Ktot % 32 == 0) || kB < Ktot;
        const int kc  = ok ? kB : 0;
        const _Float16 bv = Bs[kc * 16 + l15];
        B.h[s] = ok ? bv : (_Float16)0;
      }
    }
    acc = __builtin_amdgcn_wmma_f32_16x16x32_f16(false, A.v, false, B.v,
                                                 (short)0, acc, false, false);
  }

  const int c = (nt << 4) + l15;            // D: lane -> N
  const float bv = bias ? bias[c] : 0.f;
#pragma unroll
  for (int r = 0; r < 8; ++r) {
    const int m  = r + (hi << 3);           // D: vgpr -> M
    const int xx = x0 + m;
    const size_t pix = ((size_t)img * H + y) * W + xx;
    float v = acc[r] + bv;
    if (act == 1) v = fmaxf(v, 0.f);
    if (res32) v += res32[pix * Cout + c];
    if (out32) out32[pix * Cout + c] = v;
    if (out16) out16[pix * oStride + oOff + c] = (h16)v;
  }
}

// ================== attention: patch-correlation scores ====================
// S[q][p] = <patch_q(fd), patch_p(fd)> * invnorm[p], one image per launch.
// A from NHWC fd (channel-contiguous); B from channel-major fdT so a key tile
// is a contiguous row segment (one clamped half load per element, zeroed out of range).
__global__ __launch_bounds__(128) void attn_scores_wmma(
    const h16* __restrict__ fd, const h16* __restrict__ fdT,
    const float* __restrict__ invn, float* __restrict__ S, int img)
{
  const int L = 4096, C = 64, Hh = 64, Wh = 64, Ktot = 576;
  const int wave = (blockIdx.x * blockDim.x + threadIdx.x) >> 5;
  const int lane = threadIdx.x & 31;
  const int Ntiles = 256;
  if (wave >= 256 * Ntiles) return;
  const int mt = wave / Ntiles;
  const int nt = wave - mt * Ntiles;
  const int qy  = mt >> 2;
  const int qx0 = (mt & 3) << 4;
  const int hi  = lane >> 4, l15 = lane & 15;
  const int aqx = qx0 + l15;
  const int py  = nt >> 2;                   // key tile: row py, cols px0..px0+15
  const int px0 = (nt & 3) << 4;
  const h16* fdi  = fd  + (size_t)img * L * C;
  const h16* fdTi = fdT + (size_t)img * L * C;

  v8f acc = {0.f,0.f,0.f,0.f,0.f,0.f,0.f,0.f};

  for (int k0 = 0; k0 < Ktot; k0 += 32) {
    Frag A, B;
#pragma unroll
    for (int g = 0; g < 2; ++g) {
      const int kg  = k0 + (hi << 3) + (g << 4);
      const int tap = kg >> 6;
      const int cin = kg & 63;
      const int ky = tap / 3, kx = tap - 3 * ky;
      const int yy = qy  + ky - 1;
      const int xx = aqx + kx - 1;
      const bool ok = ((unsigned)yy < (unsigned)Hh) && ((unsigned)xx < (unsigned)Wh);
      const int yc = iclamp(yy, 0, Hh - 1);
      const int xc = iclamp(xx, 0, Wh - 1);
      const u32x4 av = *(const u32x4*)(fdi + (((size_t)yc * Wh + xc) * C + cin));
#pragma unroll
      for (int j = 0; j < 4; ++j) A.u[(g << 2) + j] = ok ? av[j] : 0u;
    }
    {
#pragma unroll
      for (int s = 0; s < 16; ++s) {
        const int k = k0 + (hi << 3) + (s & 7) + ((s >> 3) << 4);
        const int tap = k >> 6;
        const int cin = k & 63;
        const int ky = tap / 3, kx = tap - 3 * ky;
        const int yy = py + ky - 1;
        const int xx = px0 + l15 + kx - 1;
        const bool ok = ((unsigned)yy < (unsigned)Hh) && ((unsigned)xx < (unsigned)Wh);
        const int yc = iclamp(yy, 0, Hh - 1);
        const int xc = iclamp(xx, 0, Wh - 1);
        const _Float16 v = fdTi[((size_t)cin * Hh + yc) * Wh + xc];
        B.h[s] = ok ? v : (_Float16)0;
      }
    }
    acc = __builtin_amdgcn_wmma_f32_16x16x32_f16(false, A.v, false, B.v,
                                                 (short)0, acc, false, false);
  }

  const int p = (nt << 4) + l15;
  const float sc = invn[(size_t)img * L + p];
#pragma unroll
  for (int r = 0; r < 8; ++r) {
    const int q = (mt << 4) + r + (hi << 3);
    S[(size_t)q * L + p] = acc[r] * sc;
  }
}

// ================== attention: masked softmax over p =======================
__global__ __launch_bounds__(256) void attn_softmax(
    const float* __restrict__ S, const float* __restrict__ mm,
    h16* __restrict__ A, int img)
{
  const int L = 4096;
  const int q = blockIdx.x;
  const int tid = threadIdx.x;
  const float* Sr = S + (size_t)q * L;
  const float* mr = mm + (size_t)img * L;
  __shared__ float red[256];

  float mx = -1e30f;
  for (int p = tid; p < L; p += 256) {
    const float l = (mr[p] != 0.f) ? 10.f * Sr[p] : 0.f;
    mx = fmaxf(mx, l);
  }
  red[tid] = mx; __syncthreads();
  for (int s = 128; s > 0; s >>= 1) {
    if (tid < s) red[tid] = fmaxf(red[tid], red[tid + s]);
    __syncthreads();
  }
  mx = red[0]; __syncthreads();

  float sum = 0.f;
  for (int p = tid; p < L; p += 256) {
    const float l = (mr[p] != 0.f) ? 10.f * Sr[p] : 0.f;
    sum += __expf(l - mx);
  }
  red[tid] = sum; __syncthreads();
  for (int s = 128; s > 0; s >>= 1) {
    if (tid < s) red[tid] += red[tid + s];
    __syncthreads();
  }
  const float inv = 1.f / red[0];

  for (int p = tid; p < L; p += 256) {
    const float l = (mr[p] != 0.f) ? 10.f * Sr[p] : 0.f;
    const float v = __expf(l - mx) * inv * ((mr[p] != 0.f) ? 1.f : 0.f);
    A[(size_t)q * L + p] = (h16)v;
  }
}

// ======= attention: transposed-conv reconstruction as per-(dy,dx) GEMM =====
__global__ __launch_bounds__(128) void attn_deconv_wmma(
    const h16* __restrict__ Aattn, const h16* __restrict__ xf,
    float* __restrict__ out, int img, int dy, int dx)
{
  const int L = 4096, C = 64, H = 128, W = 128;
  const int wave = (blockIdx.x * blockDim.x + threadIdx.x) >> 5;
  const int lane = threadIdx.x & 31;
  if (wave >= 256 * 4) return;
  const int mt = wave >> 2;
  const int nt = wave & 3;
  const int hi = lane >> 4, l15 = lane & 15;
  const int qa = (mt << 4) + l15;

  v8f acc = {0.f,0.f,0.f,0.f,0.f,0.f,0.f,0.f};

  for (int k0 = 0; k0 < L; k0 += 32) {
    Frag A, B;
    // A: contiguous halves in the softmax row -> 2x unconditional b128
    A.q[0] = *(const u32x4*)(Aattn + ((size_t)qa * L + k0 + (hi << 3)));
    A.q[1] = *(const u32x4*)(Aattn + ((size_t)qa * L + k0 + (hi << 3) + 16));
    // B: element s -> pixel p (table order, same as A), channel (nt<<4)+l15
#pragma unroll
    for (int s = 0; s < 16; ++s) {
      const int p  = k0 + (hi << 3) + (s & 7) + ((s >> 3) << 4);
      const int py = p >> 6, px = p & 63;
      const int Y  = 2 * py + dy - 1;
      const int Xc = 2 * px + dx - 1;
      const bool ok = ((unsigned)Y < (unsigned)H) && ((unsigned)Xc < (unsigned)W);
      const size_t bbase = (((size_t)img * H + (ok ? Y : 0)) * W + (ok ? Xc : 0)) * C + (nt << 4) + l15;
      const _Float16 v = xf[bbase];
      B.h[s] = ok ? v : (_Float16)0;
    }
    acc = __builtin_amdgcn_wmma_f32_16x16x32_f16(false, A.v, false, B.v,
                                                 (short)0, acc, false, false);
  }

  const int c = (nt << 4) + l15;
#pragma unroll
  for (int r = 0; r < 8; ++r) {
    const int q  = (mt << 4) + r + (hi << 3);
    const int qy = q >> 6, qx = q & 63;
    const int Y  = 2 * qy + dy - 1;
    const int Xc = 2 * qx + dx - 1;
    if ((unsigned)Y < (unsigned)H && (unsigned)Xc < (unsigned)W) {
      const size_t idx = (((size_t)img * H + Y) * W + Xc) * C + c;
      out[idx] += 0.25f * acc[r];
    }
  }
}

// =========================== instance norm =================================
__global__ __launch_bounds__(256) void inorm_kernel(
    const float* __restrict__ in, float* __restrict__ resio,
    h16* __restrict__ out16, float* __restrict__ outNCHW, int C, int act)
{
  const int HW = 128 * 128;
  const int img = blockIdx.x / C;
  const int c   = blockIdx.x - img * C;
  const int tid = threadIdx.x;
  const size_t base = (size_t)img * HW * C + c;

  float s = 0.f, s2 = 0.f;
  for (int i = tid; i < HW; i += 256) {
    const float v = in[base + (size_t)i * C];
    s += v; s2 += v * v;
  }
  __shared__ float r1[256], r2[256];
  r1[tid] = s; r2[tid] = s2; __syncthreads();
  for (int st = 128; st > 0; st >>= 1) {
    if (tid < st) { r1[tid] += r1[tid + st]; r2[tid] += r2[tid + st]; }
    __syncthreads();
  }
  const float mu  = r1[0] * (1.f / HW);
  const float var = r2[0] * (1.f / HW) - mu * mu;
  const float rv  = rsqrtf(var + 1e-5f);

  for (int i = tid; i < HW; i += 256) {
    const size_t idx = base + (size_t)i * C;
    float v = (in[idx] - mu) * rv;
    if (act == 1)      v = fmaxf(v, 0.f);
    else if (act == 2) v = (v > 0.f) ? v : (__expf(v) - 1.f);
    if (resio) { v += resio[idx]; resio[idx] = v; }
    if (out16)   out16[idx] = (h16)v;
    if (outNCHW) outNCHW[((size_t)img * C + c) * HW + i] = v;
  }
}

// =========================== small helpers =================================
__global__ void k_nchw_to_nhwc16(const float* __restrict__ s, h16* __restrict__ d, int C) {
  const size_t idx = (size_t)blockIdx.x * blockDim.x + threadIdx.x;
  const size_t n = (size_t)4 * 16384 * C;
  if (idx >= n) return;
  const int c = idx % C;
  const size_t pix = idx / C;
  const int img = (int)(pix >> 14);
  const int i = (int)(pix & 16383);
  d[idx] = (h16)s[((size_t)img * C + c) * 16384 + i];
}
__global__ void k_nchw_to_nhwc32(const float* __restrict__ s, float* __restrict__ d, int C) {
  const size_t idx = (size_t)blockIdx.x * blockDim.x + threadIdx.x;
  const size_t n = (size_t)4 * 16384 * C;
  if (idx >= n) return;
  const int c = idx % C;
  const size_t pix = idx / C;
  const int img = (int)(pix >> 14);
  const int i = (int)(pix & 16383);
  d[idx] = s[((size_t)img * C + c) * 16384 + i];
}
// weights: f32 [nf][O][I][t] -> f16 [nf][t*I][O]
__global__ void k_wconv(const float* __restrict__ s, h16* __restrict__ d,
                        int NF, int O, int I, int taps) {
  const size_t idx = (size_t)blockIdx.x * blockDim.x + threadIdx.x;
  const size_t n = (size_t)NF * O * I * taps;
  if (idx >= n) return;
  const int t = idx % taps;
  size_t tmp = idx / taps;
  const int i = tmp % I; tmp /= I;
  const int o = tmp % O;
  const int f = (int)(tmp / O);
  d[(size_t)f * taps * I * O + ((size_t)(t * I + i)) * O + o] = (h16)s[idx];
}
__global__ void k_add16(const h16* __restrict__ a, int as, int ao,
                        const h16* __restrict__ b, int bs, int bo,
                        h16* __restrict__ o, int npix) {
  const size_t idx = (size_t)blockIdx.x * blockDim.x + threadIdx.x;
  if (idx >= (size_t)npix * 16) return;
  const size_t pix = idx >> 4;
  const int c = (int)(idx & 15);
  o[pix * 16 + c] = a[pix * as + ao + c] + b[pix * bs + bo + c];
}
__global__ void k_pack_cc(const float* __restrict__ hres, const float* __restrict__ attn,
                          const float* __restrict__ hd, h16* __restrict__ cc) {
  const size_t idx = (size_t)blockIdx.x * blockDim.x + threadIdx.x;
  if (idx >= (size_t)65536 * 64) return;
  const size_t pix = idx >> 6;
  const int c = (int)(idx & 63);
  const size_t b = pix * 192;
  cc[b + c]       = (h16)hres[idx];
  cc[b + 64 + c]  = (h16)attn[idx];
  cc[b + 128 + c] = (h16)hd[idx];
}
__global__ void k_zero32(float* __restrict__ p, size_t n) {
  const size_t idx = (size_t)blockIdx.x * blockDim.x + threadIdx.x;
  if (idx < n) p[idx] = 0.f;
}
__global__ void k_make_fd(const float* __restrict__ X, h16* __restrict__ fd) {
  const size_t idx = (size_t)blockIdx.x * blockDim.x + threadIdx.x;
  if (idx >= (size_t)4 * 4096 * 64) return;
  const int c = (int)(idx & 63);
  const int p = (int)((idx >> 6) & 4095);
  const int img = (int)(idx >> 18);
  const int qy = p >> 6, qx = p & 63;
  fd[idx] = (h16)X[(((size_t)img * 64 + c) * 128 + 2 * qy) * 128 + 2 * qx];
}
// channel-major copy of fd: fdT[img][c][qy][qx]
__global__ void k_make_fdT(const float* __restrict__ X, h16* __restrict__ fdT) {
  const size_t idx = (size_t)blockIdx.x * blockDim.x + threadIdx.x;
  if (idx >= (size_t)4 * 64 * 4096) return;
  const int qx = (int)(idx & 63);
  const int qy = (int)((idx >> 6) & 63);
  const int c  = (int)((idx >> 12) & 63);
  const int img = (int)(idx >> 18);
  fdT[idx] = (h16)X[(((size_t)img * 64 + c) * 128 + 2 * qy) * 128 + 2 * qx];
}
__global__ void k_make_mm(const float* __restrict__ mask, float* __restrict__ mm) {
  const size_t idx = (size_t)blockIdx.x * blockDim.x + threadIdx.x;
  if (idx >= (size_t)4 * 4096) return;
  const int p = (int)(idx & 4095);
  const int img = (int)(idx >> 12);
  const int qy = p >> 6, qx = p & 63;
  float s = 0.f;
  for (int ky = -1; ky <= 1; ++ky)
    for (int kx = -1; kx <= 1; ++kx) {
      const int yy = qy + ky, xx = qx + kx;
      if ((unsigned)yy < 64u && (unsigned)xx < 64u)
        s += fabsf(mask[((size_t)img * 128 + 2 * yy) * 128 + 2 * xx]);
    }
  mm[idx] = (s == 0.f) ? 1.f : 0.f;
}
__global__ void k_make_invnorm(const h16* __restrict__ fd, float* __restrict__ invn) {
  const size_t idx = (size_t)blockIdx.x * blockDim.x + threadIdx.x;
  if (idx >= (size_t)4 * 4096) return;
  const int p = (int)(idx & 4095);
  const int img = (int)(idx >> 12);
  const int qy = p >> 6, qx = p & 63;
  float s = 0.f;
  for (int ky = -1; ky <= 1; ++ky)
    for (int kx = -1; kx <= 1; ++kx) {
      const int yy = qy + ky, xx = qx + kx;
      if ((unsigned)yy < 64u && (unsigned)xx < 64u) {
        const size_t base = ((size_t)img * 4096 + (size_t)yy * 64 + xx) * 64;
        for (int c = 0; c < 64; ++c) {
          const float v = (float)fd[base + c];
          s += v * v;
        }
      }
    }
  invn[idx] = 1.f / fmaxf(sqrtf(s), 1e-4f);
}

// =============================== driver ====================================
extern "C" void kernel_launch(void* const* d_in, const int* in_sizes, int n_in,
                              void* d_out, int out_size, void* d_ws, size_t ws_size,
                              hipStream_t stream) {
  (void)in_sizes; (void)n_in; (void)out_size; (void)ws_size;
  const float* X       = (const float*)d_in[0];
  const float* mask    = (const float*)d_in[1];
  const float* res_w1  = (const float*)d_in[2];
  const float* res_b1  = (const float*)d_in[3];
  const float* res_w2  = (const float*)d_in[4];
  const float* res_b2  = (const float*)d_in[5];
  const float* dmfb_w0 = (const float*)d_in[6];
  const float* dmfb_b0 = (const float*)d_in[7];
  const float* dmfb_wd = (const float*)d_in[8];
  const float* dmfb_bd = (const float*)d_in[9];
  const float* dmfb_wf = (const float*)d_in[10];
  const float* dmfb_bf = (const float*)d_in[11];
  const float* dmfb_w1 = (const float*)d_in[12];
  const float* dmfb_b1 = (const float*)d_in[13];
  const float* cat_w1  = (const float*)d_in[14];
  const float* cat_b1  = (const float*)d_in[15];
  const float* cat_w2  = (const float*)d_in[16];
  const float* cat_b2  = (const float*)d_in[17];

  const size_t P = 65536;   // 4*128*128 pixels
  const size_t L = 4096;

  char* ws = (char*)d_ws;
  size_t off = 0;
  auto alloc = [&](size_t bytes) -> char* {
    char* p = ws + off;
    off += (bytes + 255) & ~(size_t)255;
    return p;
  };
  h16*   x16     = (h16*)  alloc(P * 64 * 2);
  h16*   hres16  = (h16*)  alloc(P * 64 * 2);
  h16*   hd16    = (h16*)  alloc(P * 64 * 2);
  float* hres32  = (float*)alloc(P * 64 * 4);
  float* hd32    = (float*)alloc(P * 64 * 4);
  float* y32     = (float*)alloc(P * 128 * 4);
  h16*   y16     = (h16*)  alloc(P * 128 * 2);
  float* attn32  = (float*)alloc(P * 64 * 4);
  h16*   fd16    = (h16*)  alloc(4 * L * 64 * 2);
  h16*   fdT16   = (h16*)  alloc(4 * L * 64 * 2);
  float* mm32    = (float*)alloc(4 * L * 4);
  float* invn    = (float*)alloc(4 * L * 4);
  float* S32     = (float*)alloc(L * L * 4);
  h16*   A16     = (h16*)  alloc(L * L * 2);
  h16*   x1_16   = (h16*)  alloc(P * 16 * 2);
  h16*   ds1     = (h16*)  alloc(P * 16 * 2);
  h16*   ds2     = (h16*)  alloc(P * 16 * 2);
  h16*   ds3     = (h16*)  alloc(P * 16 * 2);
  h16*   addt    = (h16*)  alloc(P * 16 * 2);
  h16*   fused16 = (h16*)  alloc(P * 64 * 2);
  h16*   cc16    = (h16*)  alloc(P * 192 * 2);
  h16*   res_w1h = (h16*)  alloc((size_t)8 * 36864 * 2);
  h16*   res_w2h = (h16*)  alloc((size_t)8 * 36864 * 2);
  h16*   w0h     = (h16*)  alloc((size_t)4 * 9216 * 2);
  h16*   wdh     = (h16*)  alloc((size_t)16 * 2304 * 2);
  h16*   wfh     = (h16*)  alloc((size_t)12 * 2304 * 2);
  h16*   w1h     = (h16*)  alloc((size_t)4 * 4096 * 2);
  h16*   cw1h    = (h16*)  alloc((size_t)221184 * 2);
  h16*   cw2h    = (h16*)  alloc((size_t)73728 * 2);

  auto ew = [&](size_t n) { return dim3((unsigned)((n + 255) / 256)); };

  // ---- input / weight conversion -----------------------------------------
  k_nchw_to_nhwc16<<<ew(P * 64), 256, 0, stream>>>(X, x16, 64);
  k_nchw_to_nhwc16<<<ew(P * 64), 256, 0, stream>>>(X, hres16, 64);
  k_nchw_to_nhwc16<<<ew(P * 64), 256, 0, stream>>>(X, hd16, 64);
  k_nchw_to_nhwc32<<<ew(P * 64), 256, 0, stream>>>(X, hres32, 64);
  k_nchw_to_nhwc32<<<ew(P * 64), 256, 0, stream>>>(X, hd32, 64);
  k_wconv<<<ew((size_t)8 * 36864), 256, 0, stream>>>(res_w1, res_w1h, 8, 64, 64, 9);
  k_wconv<<<ew((size_t)8 * 36864), 256, 0, stream>>>(res_w2, res_w2h, 8, 64, 64, 9);
  k_wconv<<<ew((size_t)4 * 9216), 256, 0, stream>>>(dmfb_w0, w0h, 4, 16, 64, 9);
  k_wconv<<<ew((size_t)16 * 2304), 256, 0, stream>>>(dmfb_wd, wdh, 16, 16, 16, 9);
  k_wconv<<<ew((size_t)12 * 2304), 256, 0, stream>>>(dmfb_wf, wfh, 12, 16, 16, 9);
  k_wconv<<<ew((size_t)4 * 4096), 256, 0, stream>>>(dmfb_w1, w1h, 4, 64, 64, 1);
  k_wconv<<<ew((size_t)221184), 256, 0, stream>>>(cat_w1, cw1h, 1, 128, 192, 9);
  k_wconv<<<ew((size_t)73728), 256, 0, stream>>>(cat_w2, cw2h, 1, 64, 128, 9);

  // one block = 4 M-tiles x 1 N-tile; Mtiles = 4096 (divisible by 4)
  auto conv = [&](const h16* in, const h16* w, const float* b, const float* res,
                  float* o32, h16* o16, int Cin, int Cout, int ks, int dil,
                  int act, int oStride, int oOff) {
    const int Mtiles = 4 * 128 * 8;
    dim3 grid((Mtiles / 4) * (Cout / 16));
#define CONV_CASE(CI, KSV, DI)                                                  \
    if (Cin == CI && ks == KSV && dil == DI) {                                  \
      conv_wmma<CI, KSV, DI><<<grid, 128, 0, stream>>>(                         \
          in, w, b, res, o32, o16, 4, 128, 128, Cout, act, oStride, oOff);      \
      return;                                                                   \
    }
    CONV_CASE(64, 3, 2)
    CONV_CASE(64, 3, 1)
    CONV_CASE(16, 3, 1)
    CONV_CASE(16, 3, 2)
    CONV_CASE(16, 3, 4)
    CONV_CASE(16, 3, 8)
    CONV_CASE(64, 1, 1)
    CONV_CASE(192, 3, 1)
    CONV_CASE(128, 3, 1)
#undef CONV_CASE
  };

  // ---- residual branch ----------------------------------------------------
  for (int i = 0; i < 8; ++i) {
    conv(hres16, res_w1h + (size_t)i * 36864, res_b1 + i * 64, nullptr,
         y32, nullptr, 64, 64, 3, 2, 0, 0, 0);
    inorm_kernel<<<4 * 64, 256, 0, stream>>>(y32, nullptr, y16, nullptr, 64, 1);
    conv(y16, res_w2h + (size_t)i * 36864, res_b2 + i * 64, nullptr,
         y32, nullptr, 64, 64, 3, 1, 0, 0, 0);
    inorm_kernel<<<4 * 64, 256, 0, stream>>>(y32, hres32, hres16, nullptr, 64, 0);
  }

  // ---- contextual attention ----------------------------------------------
  k_make_fd<<<ew((size_t)4 * L * 64), 256, 0, stream>>>(X, fd16);
  k_make_fdT<<<ew((size_t)4 * L * 64), 256, 0, stream>>>(X, fdT16);
  k_make_mm<<<ew((size_t)4 * L), 256, 0, stream>>>(mask, mm32);
  k_make_invnorm<<<ew((size_t)4 * L), 256, 0, stream>>>(fd16, invn);
  k_zero32<<<ew(P * 64), 256, 0, stream>>>(attn32, P * 64);
  for (int img = 0; img < 4; ++img) {
    attn_scores_wmma<<<(256 * 256 + 3) / 4, 128, 0, stream>>>(fd16, fdT16, invn, S32, img);
    attn_softmax<<<4096, 256, 0, stream>>>(S32, mm32, A16, img);
    for (int dy = 0; dy < 4; ++dy)
      for (int dx = 0; dx < 4; ++dx)
        attn_deconv_wmma<<<(256 * 4 + 3) / 4, 128, 0, stream>>>(
            A16, x16, attn32, img, dy, dx);
  }

  // ---- DMFB branch --------------------------------------------------------
  const int DILS[4] = {1, 2, 4, 8};
  for (int i = 0; i < 4; ++i) {
    conv(hd16, w0h + (size_t)i * 9216, dmfb_b0 + i * 16, nullptr,
         nullptr, x1_16, 64, 16, 3, 1, 1, 16, 0);
    conv(x1_16, wdh + (size_t)(i * 4 + 0) * 2304, dmfb_bd + (i * 4 + 0) * 16,
         nullptr, nullptr, fused16, 16, 16, 3, DILS[0], 0, 64, 0);
    conv(x1_16, wdh + (size_t)(i * 4 + 1) * 2304, dmfb_bd + (i * 4 + 1) * 16,
         nullptr, nullptr, ds1, 16, 16, 3, DILS[1], 0, 16, 0);
    conv(x1_16, wdh + (size_t)(i * 4 + 2) * 2304, dmfb_bd + (i * 4 + 2) * 16,
         nullptr, nullptr, ds2, 16, 16, 3, DILS[2], 0, 16, 0);
    conv(x1_16, wdh + (size_t)(i * 4 + 3) * 2304, dmfb_bd + (i * 4 + 3) * 16,
         nullptr, nullptr, ds3, 16, 16, 3, DILS[3], 0, 16, 0);
    k_add16<<<ew(P * 16), 256, 0, stream>>>(fused16, 64, 0, ds1, 16, 0, addt, (int)P);
    conv(addt, wfh + (size_t)(i * 3 + 0) * 2304, dmfb_bf + (i * 3 + 0) * 16,
         nullptr, nullptr, fused16, 16, 16, 3, 1, 0, 64, 16);          // t2
    k_add16<<<ew(P * 16), 256, 0, stream>>>(fused16, 64, 16, ds2, 16, 0, addt, (int)P);
    conv(addt, wfh + (size_t)(i * 3 + 1) * 2304, dmfb_bf + (i * 3 + 1) * 16,
         nullptr, nullptr, fused16, 16, 16, 3, 1, 0, 64, 32);          // t3
    k_add16<<<ew(P * 16), 256, 0, stream>>>(fused16, 64, 32, ds3, 16, 0, addt, (int)P);
    conv(addt, wfh + (size_t)(i * 3 + 2) * 2304, dmfb_bf + (i * 3 + 2) * 16,
         nullptr, nullptr, fused16, 16, 16, 3, 1, 0, 64, 48);          // t4
    conv(fused16, w1h + (size_t)i * 4096, dmfb_b1 + i * 64, hd32,
         hd32, hd16, 64, 64, 1, 1, 0, 64, 0);
  }

  // ---- concat + head ------------------------------------------------------
  k_pack_cc<<<ew(P * 64), 256, 0, stream>>>(hres32, attn32, hd32, cc16);
  conv(cc16, cw1h, cat_b1, nullptr, y32, nullptr, 192, 128, 3, 1, 0, 0, 0);
  inorm_kernel<<<4 * 128, 256, 0, stream>>>(y32, nullptr, y16, nullptr, 128, 2);
  conv(y16, cw2h, cat_b2, nullptr, y32, nullptr, 128, 64, 3, 1, 0, 0, 0);
  inorm_kernel<<<4 * 64, 256, 0, stream>>>(y32, nullptr, nullptr, (float*)d_out, 64, 2);
}
